// CustomMultiHeadAttention_29721173688579
// MI455X (gfx1250) — compile-verified, matrix operand order checked by reading, not run on hardware
//
#include <hip/hip_runtime.h>
#include <hip/hip_bf16.h>

typedef __attribute__((ext_vector_type(16))) __bf16       v16bf;
typedef __attribute__((ext_vector_type(8)))  float        v8f;
typedef __attribute__((ext_vector_type(4)))  unsigned int u32x4;
typedef unsigned short u16;

#define D_MODEL 1024
#define NHEAD   16
#define D_HEAD  64
#define BATCH   2
#define SEQ     2048
#define MTOT    (BATCH*SEQ)   // 4096

__device__ __forceinline__ u16 f32_to_bf16(float f) {
  unsigned u = __float_as_uint(f);
  unsigned r = u + 0x7FFFu + ((u >> 16) & 1u);   // round-to-nearest-even
  return (u16)(r >> 16);
}

union FragAB {          // bf16 A/B fragment: 16 elems = two contiguous 16B runs
  v16bf v;
  u32x4 q[2];
};

// ---------------------------------------------------------------------------
// fp32 -> bf16 elementwise convert (for x)
// ---------------------------------------------------------------------------
__global__ __launch_bounds__(256) void cvt_kernel(const float* __restrict__ in,
                                                  u16* __restrict__ out, int n) {
  int i = blockIdx.x * 256 + threadIdx.x;
  if (i < n) out[i] = f32_to_bf16(in[i]);
}

// ---------------------------------------------------------------------------
// fp32 [K][N] -> bf16 transposed+scaled [N][K] (weights; 32x32 LDS transpose)
// ---------------------------------------------------------------------------
__global__ __launch_bounds__(256) void cvtT_kernel(const float* __restrict__ in,
                                                   u16* __restrict__ out,
                                                   float scale) {
  __shared__ u16 tile[32][33];
  const int tx = threadIdx.x & 31;
  const int ty = threadIdx.x >> 5;          // 0..7
  const int k0 = blockIdx.y * 32;
  const int n0 = blockIdx.x * 32;
#pragma unroll
  for (int j = 0; j < 4; ++j) {
    int k = ty + j * 8;
    tile[k][tx] = f32_to_bf16(in[(size_t)(k0 + k) * D_MODEL + n0 + tx] * scale);
  }
  __syncthreads();
#pragma unroll
  for (int j = 0; j < 4; ++j) {
    int n = ty + j * 8;
    out[(size_t)(n0 + n) * D_MODEL + k0 + tx] = tile[tx][n];
  }
}

// ---------------------------------------------------------------------------
// bf16 GEMM:  C[M,N] = A[M,K] * Bt[N,K]^T + bias[N]*bscale
// MODE 0: Cf fp32, plain [M,N]
// MODE 1: Cq bf16, split-heads layout [B,H,S,Dh]      (Q,K)
// MODE 2: Cq bf16, head-transposed layout [B,H,Dh,S]  (V)
// Block tile 64x256, BK=32, double-buffered LDS, 8 waves, wave tile 32x64.
// ---------------------------------------------------------------------------
template <int MODE>
__global__ __launch_bounds__(256) void gemm_bf16_kernel(
    const u16* __restrict__ A,   // [M][K]
    const u16* __restrict__ Bt,  // [N][K] (pre-transposed weights)
    const float* __restrict__ bias, float bscale,
    float* __restrict__ Cf, u16* __restrict__ Cq,
    int M, int N, int K) {

  __shared__ __align__(16) u16 As[2][64 * 32];    // [row][k]
  __shared__ __align__(16) u16 Bs[2][256 * 32];   // [n][k]

  const int tid    = threadIdx.x;
  const int wave   = tid >> 5;
  const int lane   = tid & 31;
  const int lane16 = lane & 15;
  const int lh     = lane >> 4;
  const int wr     = wave >> 2;            // 0..1
  const int wc     = wave & 3;             // 0..3

  const int m0 = blockIdx.y * 64;
  const int n0 = blockIdx.x * 256;

  v8f acc[2][4];
#pragma unroll
  for (int mi = 0; mi < 2; ++mi)
#pragma unroll
    for (int ni = 0; ni < 4; ++ni)
      acc[mi][ni] = (v8f){0.f,0.f,0.f,0.f,0.f,0.f,0.f,0.f};

  const int arow = tid >> 2, ag = tid & 3;     // A staging: 64 rows x 4 groups

  auto stage = [&](int k0, int buf) {
    *(u32x4*)(&As[buf][arow * 32 + ag * 8]) =
        *(const u32x4*)(A + (size_t)(m0 + arow) * K + k0 + ag * 8);
#pragma unroll
    for (int rep = 0; rep < 4; ++rep) {
      int idx = tid + rep * 256;               // 0..1023
      int n = idx >> 2, g = idx & 3;
      *(u32x4*)(&Bs[buf][n * 32 + g * 8]) =
          *(const u32x4*)(Bt + (size_t)(n0 + n) * K + k0 + g * 8);
    }
  };

  const int NIT = K / 32;
  stage(0, 0);
  __syncthreads();

  for (int it = 0; it < NIT; ++it) {
    const int cur = it & 1;
    if (it + 1 < NIT) stage((it + 1) * 32, cur ^ 1);
    if (it + 2 < NIT) {                        // prefetch 2 tiles ahead (WGP scope)
      __builtin_prefetch(A + (size_t)(m0 + arow) * K + (it + 2) * 32 + ag * 8, 0, 3);
      __builtin_prefetch(Bt + (size_t)(n0 + arow * 4) * K + (it + 2) * 32, 0, 3);
    }

    FragAB fa[2], fb[4];
#pragma unroll
    for (int mi = 0; mi < 2; ++mi) {
      const u16* pA = &As[cur][(wr * 32 + mi * 16 + lane16) * 32 + lh * 8];
      fa[mi].q[0] = *(const u32x4*)(pA);
      fa[mi].q[1] = *(const u32x4*)(pA + 16);
    }
#pragma unroll
    for (int ni = 0; ni < 4; ++ni) {
      const u16* pB = &Bs[cur][(wc * 64 + ni * 16 + lane16) * 32 + lh * 8];
      fb[ni].q[0] = *(const u32x4*)(pB);
      fb[ni].q[1] = *(const u32x4*)(pB + 16);
    }
#pragma unroll
    for (int mi = 0; mi < 2; ++mi)
#pragma unroll
      for (int ni = 0; ni < 4; ++ni)
        acc[mi][ni] = __builtin_amdgcn_wmma_f32_16x16x32_bf16(
            false, fa[mi].v, false, fb[ni].v, (short)0, acc[mi][ni], false, false);
    __syncthreads();
  }

  // ---- epilogue (compile-time MODE) ----
#pragma unroll
  for (int mi = 0; mi < 2; ++mi) {
#pragma unroll
    for (int ni = 0; ni < 4; ++ni) {
      int col  = n0 + wc * 64 + ni * 16 + lane16;
      float bc = bias[col] * bscale;
      int rb   = m0 + wr * 32 + mi * 16 + lh * 8;
#pragma unroll
      for (int r = 0; r < 8; ++r) {
        int row = rb + r;
        float val = acc[mi][ni][r] + bc;
        if (MODE == 0) {
          Cf[(size_t)row * N + col] = val;
        } else if (MODE == 1) {
          int b = row >> 11, s = row & (SEQ - 1);
          int h = col >> 6,  d = col & 63;
          Cq[((((size_t)(b * NHEAD + h)) * SEQ + s) << 6) + d] = f32_to_bf16(val);
        } else {
          int b = row >> 11, s = row & (SEQ - 1);
          int h = col >> 6,  d = col & 63;
          Cq[((size_t)(b * NHEAD + h) * D_HEAD + d) * SEQ + s] = f32_to_bf16(val);
        }
      }
    }
  }
}

// ---------------------------------------------------------------------------
// Flash attention.  (softmax 1/sqrt(Dh) pre-folded into Q projection)
//   Q,K bf16 [B,H,S,Dh];  V bf16 [B,H,Dh,S]  -> O bf16 [B,S,D]
// 8 waves x 16 q rows = 128-row Q tile; key tile = 64, double-buffered.
// ---------------------------------------------------------------------------
__global__ __launch_bounds__(256) void flash_attn_kernel(
    const u16* __restrict__ Q, const u16* __restrict__ Kb,
    const u16* __restrict__ Vg, u16* __restrict__ Obf) {

  __shared__ __align__(16) u16 Ks[2][64 * 64];     // [key][dh]
  __shared__ __align__(16) u16 Vs[2][64 * 64];     // [d][key]
  __shared__ __align__(16) u16 Ps[8 * 16 * 64];    // per-wave P scratch [16][64]

  const int tid    = threadIdx.x;
  const int wave   = tid >> 5;
  const int lane   = tid & 31;
  const int lane16 = lane & 15;
  const int lh     = lane >> 4;

  const int bh = blockIdx.y;                    // b*NHEAD + h
  const int qt = blockIdx.x;                    // q row tile (128 rows)
  const size_t hb = (size_t)bh * SEQ * D_HEAD;  // base for Q/K [.,S,Dh] and V [.,Dh,S]

  // Q fragments (k = 64 -> 2 frags), loaded straight from global
  FragAB fq[2];
  {
    int qrow = qt * 128 + wave * 16 + lane16;
    const u16* qp = Q + hb + (size_t)qrow * D_HEAD;
#pragma unroll
    for (int f = 0; f < 2; ++f) {
      fq[f].q[0] = *(const u32x4*)(qp + f * 32 + lh * 8);
      fq[f].q[1] = *(const u32x4*)(qp + f * 32 + 16 + lh * 8);
    }
  }

  v8f accO[4];
#pragma unroll
  for (int di = 0; di < 4; ++di)
    accO[di] = (v8f){0.f,0.f,0.f,0.f,0.f,0.f,0.f,0.f};
  float mrow[8], lrow[8];
#pragma unroll
  for (int r = 0; r < 8; ++r) { mrow[r] = -1e30f; lrow[r] = 0.f; }

  u16* ps = Ps + wave * (16 * 64);

  auto stage = [&](int kt, int buf) {
#pragma unroll
    for (int rep = 0; rep < 2; ++rep) {
      int idx = tid + rep * 256;                // 0..511
      int row = idx >> 3, g = idx & 7;          // 64 rows x 8 groups
      *(u32x4*)(&Ks[buf][row * 64 + g * 8]) =
          *(const u32x4*)(Kb + hb + (size_t)(kt * 64 + row) * D_HEAD + g * 8);
      *(u32x4*)(&Vs[buf][row * 64 + g * 8]) =
          *(const u32x4*)(Vg + hb + (size_t)row * SEQ + kt * 64 + g * 8);
    }
  };

  const int NIT = SEQ / 64;                     // 32
  stage(0, 0);
  __syncthreads();

  for (int kt = 0; kt < NIT; ++kt) {
    const int cur = kt & 1;
    if (kt + 1 < NIT) stage(kt + 1, cur ^ 1);
    if (kt + 2 < NIT) {                         // prefetch 2 tiles ahead (WGP scope)
      __builtin_prefetch(Kb + hb + (size_t)(kt + 2) * 64 * D_HEAD + tid * 16, 0, 3);
      __builtin_prefetch(Vg + hb + (size_t)(tid >> 2) * SEQ + (kt + 2) * 64 + (tid & 3) * 16, 0, 3);
    }

    // ---- S = Q * K^T  (16 x 64 = 4 C-frags, 8 WMMA) ----
    v8f sf[4];
#pragma unroll
    for (int ni = 0; ni < 4; ++ni)
      sf[ni] = (v8f){0.f,0.f,0.f,0.f,0.f,0.f,0.f,0.f};
#pragma unroll
    for (int ni = 0; ni < 4; ++ni) {
#pragma unroll
      for (int f = 0; f < 2; ++f) {
        FragAB fk;
        const u16* pK = &Ks[cur][(ni * 16 + lane16) * 64 + f * 32 + lh * 8];
        fk.q[0] = *(const u32x4*)(pK);
        fk.q[1] = *(const u32x4*)(pK + 16);
        sf[ni] = __builtin_amdgcn_wmma_f32_16x16x32_bf16(
            false, fq[f].v, false, fk.v, (short)0, sf[ni], false, false);
      }
    }

    // ---- online softmax over 64 keys ----
    float pnew[4][8];
#pragma unroll
    for (int r = 0; r < 8; ++r) {
      float s0 = sf[0][r], s1 = sf[1][r], s2 = sf[2][r], s3 = sf[3][r];
      float rm = fmaxf(fmaxf(s0, s1), fmaxf(s2, s3));
      rm = fmaxf(rm, __shfl_xor(rm, 1, 32));
      rm = fmaxf(rm, __shfl_xor(rm, 2, 32));
      rm = fmaxf(rm, __shfl_xor(rm, 4, 32));
      rm = fmaxf(rm, __shfl_xor(rm, 8, 32));
      float mnew = fmaxf(mrow[r], rm);
      float p0 = __expf(s0 - mnew);
      float p1 = __expf(s1 - mnew);
      float p2 = __expf(s2 - mnew);
      float p3 = __expf(s3 - mnew);
      float rs = (p0 + p1) + (p2 + p3);
      rs += __shfl_xor(rs, 1, 32);
      rs += __shfl_xor(rs, 2, 32);
      rs += __shfl_xor(rs, 4, 32);
      rs += __shfl_xor(rs, 8, 32);
      float corr = __expf(mrow[r] - mnew);
      lrow[r] = lrow[r] * corr + rs;
      mrow[r] = mnew;
      pnew[0][r] = p0; pnew[1][r] = p1; pnew[2][r] = p2; pnew[3][r] = p3;
#pragma unroll
      for (int di = 0; di < 4; ++di) accO[di][r] *= corr;
    }

    // ---- C-frag -> A-frag layout change through per-wave LDS bounce ----
#pragma unroll
    for (int ni = 0; ni < 4; ++ni)
#pragma unroll
      for (int r = 0; r < 8; ++r)
        ps[(r + 8 * lh) * 64 + ni * 16 + lane16] = f32_to_bf16(pnew[ni][r]);

    FragAB pa[2];
#pragma unroll
    for (int f = 0; f < 2; ++f) {
      const u16* pp = ps + lane16 * 64 + f * 32 + lh * 8;
      pa[f].q[0] = *(const u32x4*)(pp);
      pa[f].q[1] = *(const u32x4*)(pp + 16);
    }

    // ---- O += P * V  (4 d-tiles x 2 k-frags = 8 WMMA) ----
#pragma unroll
    for (int di = 0; di < 4; ++di) {
#pragma unroll
      for (int f = 0; f < 2; ++f) {
        FragAB fv;
        const u16* vp = &Vs[cur][(di * 16 + lane16) * 64 + f * 32 + lh * 8];
        fv.q[0] = *(const u32x4*)(vp);
        fv.q[1] = *(const u32x4*)(vp + 16);
        accO[di] = __builtin_amdgcn_wmma_f32_16x16x32_bf16(
            false, pa[f].v, false, fv.v, (short)0, accO[di], false, false);
      }
    }
    __syncthreads();
  }

  // ---- normalize and write merged-head bf16 output [B,S,D] ----
  const int b = bh >> 4;
  const int h = bh & 15;
#pragma unroll
  for (int r = 0; r < 8; ++r) {
    float inv = 1.0f / lrow[r];
    int s = qt * 128 + wave * 16 + r + 8 * lh;
#pragma unroll
    for (int di = 0; di < 4; ++di) {
      int col = h * 64 + di * 16 + lane16;
      Obf[((size_t)(b * SEQ + s)) * D_MODEL + col] = f32_to_bf16(accO[di][r] * inv);
    }
  }
}

// ---------------------------------------------------------------------------
extern "C" void kernel_launch(void* const* d_in, const int* in_sizes, int n_in,
                              void* d_out, int out_size, void* d_ws, size_t ws_size,
                              hipStream_t stream) {
  (void)in_sizes; (void)n_in; (void)out_size; (void)ws_size;

  const float* x  = (const float*)d_in[0];
  const float* Wq = (const float*)d_in[1];
  const float* bq = (const float*)d_in[2];
  const float* Wk = (const float*)d_in[3];
  const float* bk = (const float*)d_in[4];
  const float* Wv = (const float*)d_in[5];
  const float* bv = (const float*)d_in[6];
  const float* Wo = (const float*)d_in[7];
  const float* bo = (const float*)d_in[8];

  const size_t xN = (size_t)MTOT * D_MODEL;     // 4M elems
  const size_t wN = (size_t)D_MODEL * D_MODEL;  // 1M elems

  char* p = (char*)d_ws;
  u16* x_bf  = (u16*)p; p += xN * 2;
  u16* wq_bf = (u16*)p; p += wN * 2;   // transposed [N][K], pre-scaled 1/8
  u16* wk_bf = (u16*)p; p += wN * 2;
  u16* wv_bf = (u16*)p; p += wN * 2;
  u16* wo_bf = (u16*)p; p += wN * 2;
  u16* Qbf   = (u16*)p; p += xN * 2;   // [B,H,S,Dh] (already * 1/sqrt(Dh))
  u16* Kbf   = (u16*)p; p += xN * 2;   // [B,H,S,Dh]
  u16* Vbf   = (u16*)p; p += xN * 2;   // [B,H,Dh,S]
  u16* Abf   = (u16*)p; p += xN * 2;   // attention out [B,S,D]

  const float qscale = 0.125f;                  // 1/sqrt(D_HEAD)

  cvt_kernel<<<dim3((unsigned)((xN + 255) / 256)), 256, 0, stream>>>(x, x_bf, (int)xN);
  dim3 gt(D_MODEL / 32, D_MODEL / 32);          // (32, 32)
  cvtT_kernel<<<gt, 256, 0, stream>>>(Wq, wq_bf, qscale);
  cvtT_kernel<<<gt, 256, 0, stream>>>(Wk, wk_bf, 1.0f);
  cvtT_kernel<<<gt, 256, 0, stream>>>(Wv, wv_bf, 1.0f);
  cvtT_kernel<<<gt, 256, 0, stream>>>(Wo, wo_bf, 1.0f);

  dim3 gg(D_MODEL / 256, MTOT / 64);            // (4, 64)
  gemm_bf16_kernel<1><<<gg, 256, 0, stream>>>(x_bf, wq_bf, bq, qscale, nullptr, Qbf,
                                              MTOT, D_MODEL, D_MODEL);
  gemm_bf16_kernel<1><<<gg, 256, 0, stream>>>(x_bf, wk_bf, bk, 1.0f, nullptr, Kbf,
                                              MTOT, D_MODEL, D_MODEL);
  gemm_bf16_kernel<2><<<gg, 256, 0, stream>>>(x_bf, wv_bf, bv, 1.0f, nullptr, Vbf,
                                              MTOT, D_MODEL, D_MODEL);

  flash_attn_kernel<<<dim3(SEQ / 128, BATCH * NHEAD), 256, 0, stream>>>(
      Qbf, Kbf, Vbf, Abf);

  gemm_bf16_kernel<0><<<gg, 256, 0, stream>>>(Abf, wo_bf, bo, 1.0f, (float*)d_out, nullptr,
                                              MTOT, D_MODEL, D_MODEL);
}
